// BiLstmCrf_27968827032186
// MI455X (gfx1250) — matrix operand order repaired and checked by reading, not compiled, not run on hardware
//
#include <hip/hip_runtime.h>
#include <hip/hip_bf16.h>

typedef __attribute__((ext_vector_type(16))) _Float16 v16h;
typedef __attribute__((ext_vector_type(8)))  float    v8f;

#define TAGS      10
#define START_TAG 8
#define END_TAG   9
#define NEGINF    (-10000.0f)
#define K2        1.44269504088896340736f   /* log2(e) */
#define LN2       0.69314718055994530942f   /* ln(2)   */

// Cross-half-wave swap (lane i <-> lane i+16) as pure VALU: v_permlanex16_b32.
__device__ __forceinline__ float swap16(float x)
{
    int xi = __builtin_bit_cast(int, x);
    int r  = __builtin_amdgcn_permlanex16(xi, xi, 0x76543210, 0xfedcba98,
                                          false, false);
    return __builtin_bit_cast(float, r);
}

// WGP-scope prefetch (scope field 0): pulls into all cache levels, stops on
// WGP hit. Address must be valid (non-speculative) -> callers clamp it.
__device__ __forceinline__ void prefetch_wgp(const void* p)
{
    asm volatile("global_prefetch_b8 %0, off" :: "v"(p));
}

// 8-way register select, l in [0,8)
__device__ __forceinline__ float sel8(int l, float2 p0, float2 p1,
                                      float2 p2, float2 p3)
{
    float eva = (l & 1) ? p0.y : p0.x;
    float evb = (l & 1) ? p1.y : p1.x;
    float evc = (l & 1) ? p2.y : p2.x;
    float evd = (l & 1) ? p3.y : p3.x;
    float eve = (l & 2) ? evb : eva;
    float evf = (l & 2) ? evd : evc;
    return (l & 4) ? evf : eve;
}

// One wave = 16 batch rows (upper half-wave mirrors the lower one; it exists
// to carry the K=16..31 of A and rows 8..15 of D for the WMMA).
// Alpha state is kept in base-2 units so the recurrence uses raw v_exp_f32 /
// v_log_f32:  acc2'[j] = em[j]*log2e + m2 + mT2 + log2( (A x B)[j] ).
__global__ void __launch_bounds__(32)
crf_fwd_kernel(const float* __restrict__ emission,
               const float* __restrict__ transitions,
               const int*   __restrict__ labels,
               const int*   __restrict__ length,
               float*       __restrict__ per_batch,
               int B, int L)
{
    __shared__ float sT[128];
    for (int i = threadIdx.x; i < 128; i += 32)
        sT[i] = (i < TAGS * TAGS) ? transitions[i] : NEGINF;
    __syncthreads();

    // global max of T (wave-uniform broadcast reads)
    float mT = sT[0];
    #pragma unroll 4
    for (int i = 1; i < TAGS * TAGS; ++i) mT = fmaxf(mT, sT[i]);
    const float mT2 = mT * K2;

    const int lane = threadIdx.x & 31;
    const int col  = lane & 15;
    const int half = lane >> 4;
    const int b    = blockIdx.x * 16 + col;
    if (blockIdx.x * 16 >= B) return;

    // ---- constant A matrix: A[M=j][K=i] = exp2((T[i][j]-mT)*log2e) ----
    v16h amat;
    #pragma unroll
    for (int e = 0; e < 16; ++e) amat[e] = (_Float16)0.0f;
    #pragma unroll
    for (int e = 0; e < 16; ++e) {
        const int i = e + 16 * half;
        float v = 0.0f;
        if (col < TAGS && i < TAGS)
            v = __builtin_amdgcn_exp2f((sT[i * TAGS + col] - mT) * K2);
        amat[e] = (_Float16)v;
    }

    v8f zero8;
    #pragma unroll
    for (int r = 0; r < 8; ++r) zero8[r] = 0.0f;

    const int len = length[b];
    int maxLen = len;
    #pragma unroll
    for (int s = 16; s > 0; s >>= 1) {
        int o = __shfl_xor(maxLen, s, 32);
        maxLen = (o > maxLen) ? o : maxLen;
    }

    const float* emB  = emission + (size_t)b * L * TAGS;
    const int*   labB = labels   + (size_t)b * L;

    // B matrix register: upper half & elements 10..15 stay zero forever.
    v16h bmat;
    #pragma unroll
    for (int e = 0; e < 16; ++e) bmat[e] = (_Float16)0.0f;

    float a2[TAGS];            // alpha * log2e (only the lower half-wave's copy is used)
    float score   = 0.0f;
    float m2state = 0.0f;      // running max of a2 (exact)
    int   prev    = START_TAG;
    int   lab_cur = 0;         // labels[b][t] for the upcoming step (pipelined)

    // ---- t = 0 exact in f32, full EXEC (START sentinel would underflow) ----
    {
        float2 q0 = *(const float2*)(emB + 0);
        float2 q1 = *(const float2*)(emB + 2);
        float2 q2 = *(const float2*)(emB + 4);
        float2 q3 = *(const float2*)(emB + 6);
        float2 q4 = *(const float2*)(emB + 8);
        float emr[TAGS] = { q0.x, q0.y, q1.x, q1.y, q2.x,
                            q2.y, q3.x, q3.y, q4.x, q4.y };
        #pragma unroll
        for (int j = 0; j < TAGS; ++j) {
            float vmax = -1e30f;
            #pragma unroll
            for (int i = 0; i < TAGS; ++i) {
                const float a0 = (i == START_TAG) ? 0.0f : NEGINF;
                vmax = fmaxf(vmax, a0 + sT[i * TAGS + j]);
            }
            float ss = 0.0f;
            #pragma unroll
            for (int i = 0; i < TAGS; ++i) {
                const float a0 = (i == START_TAG) ? 0.0f : NEGINF;
                ss += __builtin_amdgcn_exp2f((a0 + sT[i * TAGS + j] - vmax) * K2);
            }
            const float na  = emr[j] + vmax + LN2 * __builtin_amdgcn_logf(ss);
            const float ini = (j == START_TAG) ? 0.0f : NEGINF;
            a2[j] = ((0 < len) ? na : ini) * K2;
        }
        m2state = fmaxf(fmaxf(fmaxf(fmaxf(a2[0], a2[1]), fmaxf(a2[2], a2[3])),
                              fmaxf(fmaxf(a2[4], a2[5]), fmaxf(a2[6], a2[7]))),
                        fmaxf(a2[8], a2[9]));
        // gold score, t = 0 (labels in [0,8))
        const int   l  = labB[0];
        const float ev = sel8(l, q0, q1, q2, q3);
        const float tr = sT[START_TAG * TAGS + l];
        score = (0 < len) ? (ev + tr) : 0.0f;
        prev  = (0 < len) ? l : START_TAG;
        lab_cur = labB[(L > 1) ? 1 : 0];        // pipelined label for t = 1
    }

    // ---- main recurrence: one WMMA per timestep ----
    for (int t = 1; t < maxLen; ++t) {
        const float* row = emB + (size_t)t * TAGS;

        // full EXEC: issue all memory ops early (upper lanes mirror addresses,
        // so no extra lines are touched and LDS reads broadcast).
        const int tpf = (t + 8 < L) ? (t + 8) : (L - 1);
        const int tpl = (t + 64 < L) ? (t + 64) : (L - 1);
        prefetch_wgp(emB + (size_t)tpf * TAGS);
        prefetch_wgp(labB + tpl);

        float2 p0 = *(const float2*)(row + 0);
        float2 p1 = *(const float2*)(row + 2);
        float2 p2 = *(const float2*)(row + 4);
        float2 p3 = *(const float2*)(row + 6);
        float2 p4 = *(const float2*)(row + 8);
        const int tn = (t + 1 < L) ? (t + 1) : (L - 1);
        const int lab_next = labB[tn];
        const float tr = sT[prev * TAGS + lab_cur];   // LDS, issued a stage early

        // ONLY guarded section: B rows K=0..9 (upper half must stay zero)
        if (half == 0) {
            #pragma unroll
            for (int i = 0; i < TAGS; ++i)
                bmat[i] = (_Float16)__builtin_amdgcn_exp2f(a2[i] - m2state);
        }

        v8f d = __builtin_amdgcn_wmma_f32_16x16x32_f16(
            false, amat, false, bmat, (short)0, zero8, false, false);

        // full EXEC: independent work fills the WMMA->VALU hazard window
        const bool u = (t < len);
        const int  l = lab_cur;
        const float ev = sel8(l, p0, p1, p2, p3);
        score   = u ? (score + ev + tr) : score;
        prev    = u ? l : prev;
        lab_cur = lab_next;

        float em2[TAGS];
        em2[0] = p0.x * K2; em2[1] = p0.y * K2;
        em2[2] = p1.x * K2; em2[3] = p1.y * K2;
        em2[4] = p2.x * K2; em2[5] = p2.y * K2;
        em2[6] = p3.x * K2; em2[7] = p3.y * K2;
        em2[8] = p4.x * K2; em2[9] = p4.y * K2;

        // rows 8,9 of D live in the upper half-wave: pull them down (VALU)
        const float d8 = swap16(d[0]);
        const float d9 = swap16(d[1]);

        const float base = m2state + mT2;
        float na[TAGS];
        #pragma unroll
        for (int j = 0; j < 8; ++j)
            na[j] = em2[j] + base + __builtin_amdgcn_logf(d[j]);
        na[8] = em2[8] + base + __builtin_amdgcn_logf(d8);
        na[9] = em2[9] + base + __builtin_amdgcn_logf(d9);

        const float nm =
            fmaxf(fmaxf(fmaxf(fmaxf(na[0], na[1]), fmaxf(na[2], na[3])),
                        fmaxf(fmaxf(na[4], na[5]), fmaxf(na[6], na[7]))),
                  fmaxf(na[8], na[9]));
        m2state = u ? nm : m2state;
        #pragma unroll
        for (int j = 0; j < TAGS; ++j)
            a2[j] = u ? na[j] : a2[j];
    }

    // ---- logZ and output ----
    if (half == 0) {
        float z[TAGS];
        float zmax = -1e30f;
        #pragma unroll
        for (int j = 0; j < TAGS; ++j) {
            z[j] = a2[j] + sT[j * TAGS + END_TAG] * K2;
            zmax = fmaxf(zmax, z[j]);
        }
        float zs = 0.0f;
        #pragma unroll
        for (int j = 0; j < TAGS; ++j)
            zs += __builtin_amdgcn_exp2f(z[j] - zmax);
        const float logz = (zmax + __builtin_amdgcn_logf(zs)) * LN2;
        per_batch[b] = logz - (score + sT[prev * TAGS + END_TAG]);
    }
}

// Deterministic final reduction: fixed thread count, fixed order.
__global__ void __launch_bounds__(256)
reduce_kernel(const float* __restrict__ in, float* __restrict__ out, int n)
{
    __shared__ float s[256];
    float a = 0.0f;
    for (int i = threadIdx.x; i < n; i += 256) a += in[i];
    s[threadIdx.x] = a;
    __syncthreads();
    #pragma unroll
    for (int off = 128; off > 0; off >>= 1) {
        if ((int)threadIdx.x < off) s[threadIdx.x] += s[threadIdx.x + off];
        __syncthreads();
    }
    if (threadIdx.x == 0) out[0] = s[0];
}

extern "C" void kernel_launch(void* const* d_in, const int* in_sizes, int n_in,
                              void* d_out, int out_size, void* d_ws, size_t ws_size,
                              hipStream_t stream)
{
    const float* emission    = (const float*)d_in[0];
    const float* transitions = (const float*)d_in[1];
    const int*   labels      = (const int*)d_in[2];
    const int*   length      = (const int*)d_in[3];
    float*       out         = (float*)d_out;
    float*       per_batch   = (float*)d_ws;

    const int B = in_sizes[3];                 // 8192
    const int L = in_sizes[2] / B;             // 512
    (void)n_in; (void)out_size; (void)ws_size;

    // one wave (16 batch rows) per workgroup -> 512 blocks spread over WGPs
    const int blocks = (B + 15) / 16;
    crf_fwd_kernel<<<blocks, 32, 0, stream>>>(emission, transitions, labels,
                                              length, per_batch, B, L);
    reduce_kernel<<<1, 256, 0, stream>>>(per_batch, out, B);
}
